// Attention_3762391351400
// MI455X (gfx1250) — matrix operand order repaired and checked by reading, not compiled, not run on hardware
//
#include <hip/hip_runtime.h>
#include <stdint.h>

typedef _Float16 v8h  __attribute__((ext_vector_type(8)));
typedef _Float16 v16h __attribute__((ext_vector_type(16)));
typedef float    v8f  __attribute__((ext_vector_type(8)));
typedef int      v4i  __attribute__((ext_vector_type(4)));

#define DIMV   768
#define HEADS  12
#define HD     64
#define BATCH  4
#define SEQ    2048
#define MTOT   (BATCH*SEQ)          // 8192
#define NQKV   (3*DIMV)             // 2304
#define KSTEPS (DIMV/32)            // 24

// ---------------------------------------------------------------- async copy helpers
#if __has_builtin(__builtin_amdgcn_global_load_async_to_lds_b128)
__device__ __forceinline__ void cp16(void* l, const void* g) {
    __builtin_amdgcn_global_load_async_to_lds_b128(
        (__attribute__((address_space(1))) v4i*)(g),
        (__attribute__((address_space(3))) v4i*)(l), 0, 0);
}
__device__ __forceinline__ void cp_wait() {
#if __has_builtin(__builtin_amdgcn_s_wait_asynccnt)
    __builtin_amdgcn_s_wait_asynccnt(0);
#else
    asm volatile("s_wait_asynccnt 0x0" ::: "memory");
#endif
}
#else
__device__ __forceinline__ void cp16(void* l, const void* g) {
    *(v8h*)l = *(const v8h*)g;
}
__device__ __forceinline__ void cp_wait() {}
#endif

// ---------------------------------------------------------------- WMMA helpers
__device__ __forceinline__ v8f wmma16(v16h a, v16h b, v8f c) {
    return __builtin_amdgcn_wmma_f32_16x16x32_f16(false, a, false, b, (short)0, c, false, false);
}

// A fragment: 16x32 f16 tile, row-major, row stride ld (halves).
__device__ __forceinline__ v16h load_afrag(const _Float16* p, int ld, int lane) {
    const _Float16* q = p + (size_t)(lane & 15) * ld + ((lane >> 4) << 3);
    v8h lo = *(const v8h*)(q);
    v8h hi = *(const v8h*)(q + 16);
    return __builtin_shufflevector(lo, hi, 0,1,2,3,4,5,6,7,8,9,10,11,12,13,14,15);
}

// B fragment: 32x16 with K contiguous per output column (tile stored [col][k]).
__device__ __forceinline__ v16h load_bfrag(const _Float16* p, int ld, int lane) {
    const _Float16* q = p + (size_t)(lane & 15) * ld + ((lane >> 4) << 3);
    v8h lo = *(const v8h*)(q);
    v8h hi = *(const v8h*)(q + 16);
    return __builtin_shufflevector(lo, hi, 0,1,2,3,4,5,6,7,8,9,10,11,12,13,14,15);
}

// stage a 128x32 f16 tile (row stride DIMV in global) into LDS [128][32]
__device__ __forceinline__ void stage_gemm_tile(_Float16* dst, const _Float16* src, int tid) {
    #pragma unroll
    for (int t = 0; t < 4; t++) {
        int s = tid + t * 128;
        int r = s >> 2, c = (s & 3) << 3;
        cp16(dst + r * 32 + c, src + (size_t)r * DIMV + c);
    }
}
// stage a 64x64 f16 tile (row stride srcLd) into LDS [64][64]
__device__ __forceinline__ void stage_64x64(_Float16* dst, const _Float16* src, size_t srcLd, int tid) {
    #pragma unroll
    for (int t = 0; t < 4; t++) {
        int s = tid + t * 128;
        int r = s >> 3, c = (s & 7) << 3;
        cp16(dst + r * 64 + c, src + (size_t)r * srcLd + c);
    }
}

// ---------------------------------------------------------------- fp32 -> f16
__global__ void cvt_f32_f16(const float* __restrict__ s, _Float16* __restrict__ d, int n) {
    int i = blockIdx.x * blockDim.x + threadIdx.x;
    int st = gridDim.x * blockDim.x;
    for (; i < n; i += st) d[i] = (_Float16)s[i];
}

// ---------------------------------------------------------------- QKV GEMM
// C[8192,2304] = Xh @ Wh^T + bias; Q,K scattered to [b,h,n,64]; V scattered TRANSPOSED to [b,h,64,n]
__global__ void __launch_bounds__(128) qkv_gemm(const _Float16* __restrict__ Xh,
                                                const _Float16* __restrict__ Wh,
                                                const float*    __restrict__ bias,
                                                _Float16* __restrict__ Qo,
                                                _Float16* __restrict__ Ko,
                                                _Float16* __restrict__ Vto) {
    __shared__ __align__(32) _Float16 Asl[2][128][32];
    __shared__ __align__(32) _Float16 Bsl[2][128][32];
    const int tid = threadIdx.x, lane = tid & 31, wave = tid >> 5;
    const int wm  = (wave >> 1) * 64, wn = (wave & 1) * 64;
    const int mB  = blockIdx.y * 128;
    const int nB  = blockIdx.x * 128;
    const _Float16* Abase = Xh + (size_t)mB * DIMV;
    const _Float16* Bbase = Wh + (size_t)nB * DIMV;

    stage_gemm_tile(&Asl[0][0][0], Abase, tid);
    stage_gemm_tile(&Bsl[0][0][0], Bbase, tid);
    cp_wait();
    __syncthreads();

    v8f acc[4][4] = {};
    for (int kt = 0; kt < KSTEPS; kt++) {
        const int cur = kt & 1;
        if (kt + 1 < KSTEPS) {
            stage_gemm_tile(&Asl[cur ^ 1][0][0], Abase + (size_t)(kt + 1) * 32, tid);
            stage_gemm_tile(&Bsl[cur ^ 1][0][0], Bbase + (size_t)(kt + 1) * 32, tid);
        }
        v16h a[4];
        #pragma unroll
        for (int i = 0; i < 4; i++) a[i] = load_afrag(&Asl[cur][wm + i * 16][0], 32, lane);
        #pragma unroll
        for (int j = 0; j < 4; j++) {
            v16h b = load_bfrag(&Bsl[cur][wn + j * 16][0], 32, lane);
            #pragma unroll
            for (int i = 0; i < 4; i++) acc[i][j] = wmma16(a[i], b, acc[i][j]);
        }
        cp_wait();
        __syncthreads();
    }

    #pragma unroll
    for (int j = 0; j < 4; j++) {
        int col   = nB + wn + j * 16 + (lane & 15);
        float bv  = bias[col];
        int which = col / DIMV;
        int rem   = col - which * DIMV;
        int h = rem >> 6, dd = rem & 63;
        #pragma unroll
        for (int i = 0; i < 4; i++) {
            #pragma unroll
            for (int v = 0; v < 8; v++) {
                int m  = mB + wm + i * 16 + (lane >> 4) * 8 + v;
                int bb = m >> 11, nn = m & 2047;
                _Float16 val = (_Float16)(acc[i][j][v] + bv);
                if (which == 0)      Qo [(((size_t)bb * HEADS + h) * SEQ + nn) * HD + dd] = val;
                else if (which == 1) Ko [(((size_t)bb * HEADS + h) * SEQ + nn) * HD + dd] = val;
                else                 Vto[(((size_t)bb * HEADS + h) * HD + dd) * SEQ + nn] = val;
            }
        }
    }
}

// ---------------------------------------------------------------- Flash attention
// grid = (B*H, SEQ/64); block = 128 (4 waves). Wave w owns 16 Q rows. V comes pre-transposed.
__global__ void __launch_bounds__(128) flash_attn(const _Float16* __restrict__ Q,
                                                  const _Float16* __restrict__ K,
                                                  const _Float16* __restrict__ Vt,
                                                  _Float16* __restrict__ Ao) {
    __shared__ __align__(32) _Float16 Ksl[2][64][64];
    __shared__ __align__(32) _Float16 Vtl[2][64][64];
    __shared__ __align__(32) _Float16 Psl[64][64];
    const int bh = blockIdx.x, rb = blockIdx.y;
    const int tid = threadIdx.x, lane = tid & 31, wave = tid >> 5;
    const _Float16* Qp  = Q  + (size_t)bh * SEQ * HD;
    const _Float16* Kp  = K  + (size_t)bh * SEQ * HD;
    const _Float16* Vtp = Vt + (size_t)bh * HD * SEQ;
    const int qr = rb * 64 + wave * 16;
    const float SC = 0.03608439182435161f;   // 768^-0.5 (full-dim scale, per reference)

    v16h qf[2];
    {
        const _Float16* p0 = Qp + (size_t)(qr + (lane & 15)) * HD + ((lane >> 4) << 3);
        #pragma unroll
        for (int f = 0; f < 2; f++) {
            v8h lo = *(const v8h*)(p0 + f * 32);
            v8h hi = *(const v8h*)(p0 + f * 32 + 16);
            qf[f] = __builtin_shufflevector(lo, hi, 0,1,2,3,4,5,6,7,8,9,10,11,12,13,14,15);
        }
    }

    float mrow[8], lrow[8];
    #pragma unroll
    for (int v = 0; v < 8; v++) { mrow[v] = -1e30f; lrow[v] = 0.0f; }
    v8f o[4] = {};

    stage_64x64(&Ksl[0][0][0], Kp, 64, tid);
    stage_64x64(&Vtl[0][0][0], Vtp, SEQ, tid);
    cp_wait();
    __syncthreads();

    for (int it = 0; it < SEQ / 64; it++) {
        const int cur = it & 1;
        if (it + 1 < SEQ / 64) {
            stage_64x64(&Ksl[cur ^ 1][0][0], Kp + (size_t)(it + 1) * 64 * HD, 64, tid);
            stage_64x64(&Vtl[cur ^ 1][0][0], Vtp + (size_t)(it + 1) * 64, SEQ, tid);
        }

        // S = Q @ K^T
        v8f sf[4];
        #pragma unroll
        for (int jt = 0; jt < 4; jt++) {
            v16h b0 = load_bfrag(&Ksl[cur][jt * 16][0],  64, lane);
            v16h b1 = load_bfrag(&Ksl[cur][jt * 16][32], 64, lane);
            v8f s = {};
            s = wmma16(qf[0], b0, s);
            s = wmma16(qf[1], b1, s);
            #pragma unroll
            for (int v = 0; v < 8; v++) sf[jt][v] = s[v] * SC;
        }

        // online softmax (rows in 16-lane groups: xor masks 1,2,4,8)
        float mnew[8];
        #pragma unroll
        for (int v = 0; v < 8; v++) {
            float t = fmaxf(fmaxf(sf[0][v], sf[1][v]), fmaxf(sf[2][v], sf[3][v]));
            #pragma unroll
            for (int mk = 1; mk < 16; mk <<= 1) t = fmaxf(t, __shfl_xor(t, mk, 32));
            mnew[v] = fmaxf(mrow[v], t);
        }
        float rs[8] = {0,0,0,0,0,0,0,0};
        #pragma unroll
        for (int jt = 0; jt < 4; jt++)
        #pragma unroll
        for (int v = 0; v < 8; v++) {
            float p = __expf(sf[jt][v] - mnew[v]);
            rs[v] += p;
            Psl[wave * 16 + (lane >> 4) * 8 + v][jt * 16 + (lane & 15)] = (_Float16)p;
        }
        #pragma unroll
        for (int v = 0; v < 8; v++) {
            #pragma unroll
            for (int mk = 1; mk < 16; mk <<= 1) rs[v] += __shfl_xor(rs[v], mk, 32);
            float sc = __expf(mrow[v] - mnew[v]);
            lrow[v] = lrow[v] * sc + rs[v];
            mrow[v] = mnew[v];
            #pragma unroll
            for (int dt = 0; dt < 4; dt++) o[dt][v] *= sc;
        }
        __syncthreads();

        // O += P @ V
        #pragma unroll
        for (int ks = 0; ks < 2; ks++) {
            v16h ap = load_afrag(&Psl[wave * 16][ks * 32], 64, lane);
            #pragma unroll
            for (int dt = 0; dt < 4; dt++) {
                v16h bv = load_bfrag(&Vtl[cur][dt * 16][ks * 32], 64, lane);
                o[dt] = wmma16(ap, bv, o[dt]);
            }
        }
        cp_wait();
        __syncthreads();
    }

    const int b_ = bh / HEADS, h_ = bh % HEADS;
    #pragma unroll
    for (int dt = 0; dt < 4; dt++) {
        int dcol = h_ * HD + dt * 16 + (lane & 15);
        #pragma unroll
        for (int v = 0; v < 8; v++) {
            int n_ = qr + (lane >> 4) * 8 + v;
            Ao[((size_t)b_ * SEQ + n_) * DIMV + dcol] = (_Float16)(o[dt][v] / lrow[v]);
        }
    }
}

// ---------------------------------------------------------------- Output projection
__global__ void __launch_bounds__(128) out_gemm(const _Float16* __restrict__ Ah,
                                                const _Float16* __restrict__ Wh,
                                                const float*    __restrict__ bias,
                                                float* __restrict__ out) {
    __shared__ __align__(32) _Float16 Asl[2][128][32];
    __shared__ __align__(32) _Float16 Bsl[2][128][32];
    const int tid = threadIdx.x, lane = tid & 31, wave = tid >> 5;
    const int wm  = (wave >> 1) * 64, wn = (wave & 1) * 64;
    const int mB  = blockIdx.y * 128;
    const int nB  = blockIdx.x * 128;
    const _Float16* Abase = Ah + (size_t)mB * DIMV;
    const _Float16* Bbase = Wh + (size_t)nB * DIMV;

    stage_gemm_tile(&Asl[0][0][0], Abase, tid);
    stage_gemm_tile(&Bsl[0][0][0], Bbase, tid);
    cp_wait();
    __syncthreads();

    v8f acc[4][4] = {};
    for (int kt = 0; kt < KSTEPS; kt++) {
        const int cur = kt & 1;
        if (kt + 1 < KSTEPS) {
            stage_gemm_tile(&Asl[cur ^ 1][0][0], Abase + (size_t)(kt + 1) * 32, tid);
            stage_gemm_tile(&Bsl[cur ^ 1][0][0], Bbase + (size_t)(kt + 1) * 32, tid);
        }
        v16h a[4];
        #pragma unroll
        for (int i = 0; i < 4; i++) a[i] = load_afrag(&Asl[cur][wm + i * 16][0], 32, lane);
        #pragma unroll
        for (int j = 0; j < 4; j++) {
            v16h b = load_bfrag(&Bsl[cur][wn + j * 16][0], 32, lane);
            #pragma unroll
            for (int i = 0; i < 4; i++) acc[i][j] = wmma16(a[i], b, acc[i][j]);
        }
        cp_wait();
        __syncthreads();
    }

    #pragma unroll
    for (int j = 0; j < 4; j++) {
        int col  = nB + wn + j * 16 + (lane & 15);
        float bv = bias[col];
        #pragma unroll
        for (int i = 0; i < 4; i++) {
            #pragma unroll
            for (int v = 0; v < 8; v++) {
                int m = mB + wm + i * 16 + (lane >> 4) * 8 + v;
                out[(size_t)m * DIMV + col] = acc[i][j][v] + bv;
            }
        }
    }
}

// ---------------------------------------------------------------- launch
extern "C" void kernel_launch(void* const* d_in, const int* in_sizes, int n_in,
                              void* d_out, int out_size, void* d_ws, size_t ws_size,
                              hipStream_t stream) {
    const float* x     = (const float*)d_in[0];
    const float* w_qkv = (const float*)d_in[1];
    const float* b_qkv = (const float*)d_in[2];
    const float* w_out = (const float*)d_in[3];
    const float* b_out = (const float*)d_in[4];
    float* out = (float*)d_out;

    _Float16* ws = (_Float16*)d_ws;
    size_t off = 0;
    _Float16* Xh    = ws + off; off += (size_t)MTOT * DIMV;
    _Float16* Wqkvh = ws + off; off += (size_t)NQKV * DIMV;
    _Float16* Wouth = ws + off; off += (size_t)DIMV * DIMV;
    _Float16* Qb    = ws + off; off += (size_t)BATCH * HEADS * SEQ * HD;
    _Float16* Kb    = ws + off; off += (size_t)BATCH * HEADS * SEQ * HD;
    _Float16* Vtb   = ws + off; off += (size_t)BATCH * HEADS * SEQ * HD;   // [b,h,64,n]
    _Float16* Ah    = ws + off; off += (size_t)MTOT * DIMV;

    cvt_f32_f16<<<512, 256, 0, stream>>>(x,     Xh,    MTOT * DIMV);
    cvt_f32_f16<<<256, 256, 0, stream>>>(w_qkv, Wqkvh, NQKV * DIMV);
    cvt_f32_f16<<<128, 256, 0, stream>>>(w_out, Wouth, DIMV * DIMV);

    qkv_gemm<<<dim3(NQKV / 128, MTOT / 128), 128, 0, stream>>>(Xh, Wqkvh, b_qkv, Qb, Kb, Vtb);
    flash_attn<<<dim3(BATCH * HEADS, SEQ / 64), 128, 0, stream>>>(Qb, Kb, Vtb, Ah);
    out_gemm<<<dim3(DIMV / 128, MTOT / 128), 128, 0, stream>>>(Ah, Wouth, b_out, out);
}
